// NostARHead_35150012350986
// MI455X (gfx1250) — matrix operand order repaired and checked by reading, not compiled, not run on hardware
//
#include <hip/hip_runtime.h>
#include <hip/hip_bf16.h>

// ---------------- constants (problem shape is fixed) ----------------
#define BATCH   8
#define SEQ     2048
#define DIM     2048
#define NHEAD   16
#define HDIM    128
#define LAYERS  2
#define DCAT    (LAYERS * DIM)      // 4096
#define IDIM    (4 * DCAT)          // 16384
#define PAD_ID  50257
#define EPS_LN  1e-5f
#define MTOK    (BATCH * SEQ)       // 16384 rows in the big GEMMs

typedef __attribute__((ext_vector_type(16))) __bf16 v16bf;
typedef __attribute__((ext_vector_type(8)))  __bf16 v8bf;
typedef __attribute__((ext_vector_type(8)))  float  v8f;

// ---------------- bf16 helpers (bit-exact RNE, no reliance on __bf16 arith) --
static __device__ __forceinline__ __bf16 f2bf(float f) {
  unsigned u = __builtin_bit_cast(unsigned, f);
  unsigned r = u + 0x7FFFu + ((u >> 16) & 1u);
  unsigned short h = (unsigned short)(r >> 16);
  return __builtin_bit_cast(__bf16, h);
}
static __device__ __forceinline__ float bf2f(__bf16 x) {
  unsigned short h = __builtin_bit_cast(unsigned short, x);
  unsigned u = ((unsigned)h) << 16;
  return __builtin_bit_cast(float, u);
}

// ---------------- 1) last non-pad index per batch ----------------
__global__ void idx_kernel(const int* __restrict__ ids, int* __restrict__ idx) {
  __shared__ int red[256];
  const int b = blockIdx.x;
  int best = 0;  // argmax(cumsum(mask)) == last i with mask==1, or 0 if none
  for (int i = threadIdx.x; i < SEQ; i += 256)
    if (ids[b * SEQ + i] != PAD_ID) best = i;   // i increases -> keeps max
  red[threadIdx.x] = best;
  __syncthreads();
  for (int st = 128; st > 0; st >>= 1) {
    if (threadIdx.x < st) red[threadIdx.x] = max(red[threadIdx.x], red[threadIdx.x + st]);
    __syncthreads();
  }
  if (threadIdx.x == 0) idx[b] = red[0];
}

// ---------------- 2) LayerNorm token -> bf16 ----------------
__global__ void ln_kernel(const float* __restrict__ hs, const float* __restrict__ g,
                          const float* __restrict__ bt, __bf16* __restrict__ hout) {
  const int tok = blockIdx.x;                 // 0 .. MTOK-1
  const float* x = hs + (size_t)tok * DIM;
  __shared__ float rs[256], rs2[256];
  float s = 0.f, s2 = 0.f;
  for (int i = threadIdx.x; i < DIM; i += 256) { float v = x[i]; s += v; s2 += v * v; }
  rs[threadIdx.x] = s; rs2[threadIdx.x] = s2;
  __syncthreads();
  for (int st = 128; st > 0; st >>= 1) {
    if (threadIdx.x < st) { rs[threadIdx.x] += rs[threadIdx.x + st]; rs2[threadIdx.x] += rs2[threadIdx.x + st]; }
    __syncthreads();
  }
  const float mean = rs[0] * (1.f / DIM);
  const float var  = rs2[0] * (1.f / DIM) - mean * mean;
  const float inv  = rsqrtf(var + EPS_LN);
  __bf16* out = hout + (size_t)tok * DIM;
  for (int i = threadIdx.x; i < DIM; i += 256)
    out[i] = f2bf((x[i] - mean) * inv * g[i] + bt[i]);
}

// ---------------- 3) pack weight f32 [K,N] -> bf16 WMMA B-fragment layout ----
// Layout: Wp[(((kb*2+g)*DIM) + n)*16 + t] = W[kb*32 + (t>>3)*16 + g*8 + (t&7), n]
// so each lane's B fragment is one contiguous 32-byte run.
__global__ void pack_w_kernel(const float* __restrict__ W, __bf16* __restrict__ Wp) {
  const int t = blockIdx.x * 256 + threadIdx.x;   // over DIM*DIM
  const int n = t & (DIM - 1);
  const int k = t / DIM;
  const int kb = k >> 5, r = k & 31, g = (r >> 3) & 1, tt = (r & 7) | ((r >> 4) << 3);
  Wp[((size_t)((kb * 2 + g) * DIM + n)) * 16 + tt] = f2bf(W[(size_t)k * DIM + n]);
}

// ---------------- 4) bf16 WMMA GEMM: C[MTOK,DIM] = A[MTOK,DIM] * Wp ----------
// One wave computes a 16(M) x 128(N) tile. Pointer-bumped addressing with
// immediate offsets (j*512B for the 8 B fragments; +64B / +128KiB strides per
// k-step) plus a one-deep software pipeline: the next k-step's 10 loads are in
// flight while the current step's 8 WMMAs execute, so the backend can use
// partial s_wait_loadcnt instead of draining to zero before every WMMA.
__global__ void __launch_bounds__(256) gemm_bf16_kernel(const __bf16* __restrict__ A,
                                                        const __bf16* __restrict__ Wp,
                                                        float* __restrict__ C) {
  const int wave = threadIdx.x >> 5;
  const int lane = threadIdx.x & 31;
  const int tile = blockIdx.x * 8 + wave;       // 16384 tiles total
  const int mTile = tile & 1023;                // MTOK/16 = 1024
  const int nTile = tile >> 10;                 // DIM/128 = 16
  const int g  = lane >> 4;                     // lane group (ISA K-half select)
  const int ln = lane & 15;
  const int row = mTile * 16 + ln;
  const int n0 = nTile * 128;

  // A fragment per ISA: lane group g holds K = kb*32 + {g*8..+7, 16+g*8..+7}
  const __bf16* aptr = A + (size_t)row * DIM + g * 8;
  // B fragment base for j=0; fragment j is at +j*256 elements (512B)
  const __bf16* bptr = Wp + ((size_t)(g * DIM + n0 + ln)) * 16;

  v8f acc[8];
#pragma unroll
  for (int j = 0; j < 8; ++j)
#pragma unroll
    for (int r = 0; r < 8; ++r) acc[j][r] = 0.f;

  // ---- pipeline prologue: fragments for kb = 0 ----
  v8bf alo = *(const v8bf*)(aptr);
  v8bf ahi = *(const v8bf*)(aptr + 16);
  v16bf bcur[8];
#pragma unroll
  for (int j = 0; j < 8; ++j) bcur[j] = *(const v16bf*)(bptr + j * 256);

  for (int kb = 0; kb < DIM / 32 - 1; ++kb) {
    aptr += 32;                 // next 32-K block of this A row (64 bytes)
    bptr += 2 * DIM * 16;       // next 32-K block of packed B (128 KiB)

    // issue next k-step's loads; they overlap the 8 WMMAs below
    const v8bf alo_n = *(const v8bf*)(aptr);
    const v8bf ahi_n = *(const v8bf*)(aptr + 16);
    v16bf bnext[8];
#pragma unroll
    for (int j = 0; j < 8; ++j) bnext[j] = *(const v16bf*)(bptr + j * 256);

    v16bf a;
#pragma unroll
    for (int i = 0; i < 8; ++i) { a[i] = alo[i]; a[8 + i] = ahi[i]; }
#pragma unroll
    for (int j = 0; j < 8; ++j)
      acc[j] = __builtin_amdgcn_wmma_f32_16x16x32_bf16(
          /*neg_a=*/false, a, /*neg_b=*/false, bcur[j],
          /*c_mod=*/(short)0, acc[j], /*reuse_a=*/false, /*reuse_b=*/false);

    alo = alo_n; ahi = ahi_n;
#pragma unroll
    for (int j = 0; j < 8; ++j) bcur[j] = bnext[j];
  }
  // ---- pipeline epilogue: last k-step ----
  {
    v16bf a;
#pragma unroll
    for (int i = 0; i < 8; ++i) { a[i] = alo[i]; a[8 + i] = ahi[i]; }
#pragma unroll
    for (int j = 0; j < 8; ++j)
      acc[j] = __builtin_amdgcn_wmma_f32_16x16x32_bf16(
          false, a, false, bcur[j], (short)0, acc[j], false, false);
  }

  // C/D layout: VGPR r, lane -> (m = r + 8*g, n = ln)
#pragma unroll
  for (int j = 0; j < 8; ++j) {
    const int n = n0 + j * 16 + ln;
#pragma unroll
    for (int r = 0; r < 8; ++r) {
      const int m = mTile * 16 + r + 8 * g;
      C[(size_t)m * DIM + n] = acc[j][r];
    }
  }
}

// ---------------- 5) q[b,:] = h[b, idx[b], :] @ Wq ----------------
__global__ void q_kernel(const __bf16* __restrict__ h, const int* __restrict__ idx,
                         const float* __restrict__ Wq, float* __restrict__ q) {
  const int b = blockIdx.y;
  const int n = blockIdx.x * 256 + threadIdx.x;
  __shared__ float lh[DIM];
  const __bf16* hr = h + ((size_t)(b * SEQ + idx[b])) * DIM;
  for (int k = threadIdx.x; k < DIM; k += 256) lh[k] = bf2f(hr[k]);
  __syncthreads();
  float acc = 0.f;
  for (int k = 0; k < DIM; ++k) acc = fmaf(lh[k], Wq[(size_t)k * DIM + n], acc);
  q[(size_t)b * DIM + n] = acc;
}

// ---------------- 6) scores[b,h,s] = q[b,h,:] . K[b,s,h,:] (no 1/sqrt(hd)) ---
__global__ void scores_kernel(const float* __restrict__ q, const float* __restrict__ Kf,
                              float* __restrict__ sc) {
  const int bh = blockIdx.y;
  const int b = bh / NHEAD, hh = bh % NHEAD;
  const int s = blockIdx.x * 256 + threadIdx.x;
  __shared__ float qs[HDIM];
  if (threadIdx.x < HDIM) qs[threadIdx.x] = q[(size_t)b * DIM + hh * HDIM + threadIdx.x];
  __syncthreads();
  const float4* kr = (const float4*)(Kf + ((size_t)(b * SEQ + s)) * DIM + hh * HDIM);
  float acc = 0.f;
#pragma unroll 4
  for (int d = 0; d < HDIM / 4; ++d) {
    const float4 kv = kr[d];
    acc = fmaf(qs[4 * d + 0], kv.x, acc);
    acc = fmaf(qs[4 * d + 1], kv.y, acc);
    acc = fmaf(qs[4 * d + 2], kv.z, acc);
    acc = fmaf(qs[4 * d + 3], kv.w, acc);
  }
  sc[(size_t)bh * SEQ + s] = acc;
}

// ---------------- 7) softmax over S per (b,h) ----------------
__global__ void softmax_kernel(float* __restrict__ sc) {
  float* row = sc + (size_t)blockIdx.x * SEQ;
  __shared__ float red[256];
  float m = -3.0e38f;
  for (int i = threadIdx.x; i < SEQ; i += 256) m = fmaxf(m, row[i]);
  red[threadIdx.x] = m; __syncthreads();
  for (int st = 128; st > 0; st >>= 1) {
    if (threadIdx.x < st) red[threadIdx.x] = fmaxf(red[threadIdx.x], red[threadIdx.x + st]);
    __syncthreads();
  }
  const float mx = red[0];
  __syncthreads();
  float s = 0.f;
  for (int i = threadIdx.x; i < SEQ; i += 256) { const float e = expf(row[i] - mx); row[i] = e; s += e; }
  red[threadIdx.x] = s; __syncthreads();
  for (int st = 128; st > 0; st >>= 1) {
    if (threadIdx.x < st) red[threadIdx.x] += red[threadIdx.x + st];
    __syncthreads();
  }
  const float inv = 1.f / red[0];
  __syncthreads();
  for (int i = threadIdx.x; i < SEQ; i += 256) row[i] *= inv;
}

// ---------------- 8) o[b,h,d] = sum_s w[b,h,s] * V[b,s,h,d] ----------------
__global__ void av_kernel(const float* __restrict__ sc, const float* __restrict__ Vf,
                          float* __restrict__ o) {
  const int bh = blockIdx.x;
  const int b = bh / NHEAD, hh = bh % NHEAD;
  const int d = threadIdx.x;                    // 128 threads
  const float* w = sc + (size_t)bh * SEQ;
  float acc = 0.f;
  for (int s = 0; s < SEQ; ++s)
    acc = fmaf(w[s], Vf[((size_t)(b * SEQ + s)) * DIM + hh * HDIM + d], acc);
  o[(size_t)b * DIM + hh * HDIM + d] = acc;
}

// ---------------- 9) out-proj: hcat[b, l*D + n] = o[b,:] @ Wo + bo ----------
__global__ void outproj_kernel(const float* __restrict__ o, const float* __restrict__ Wo,
                               const float* __restrict__ bo, float* __restrict__ outp) {
  const int b = blockIdx.y;
  const int n = blockIdx.x * 256 + threadIdx.x;
  __shared__ float ol[DIM];
  for (int k = threadIdx.x; k < DIM; k += 256) ol[k] = o[(size_t)b * DIM + k];
  __syncthreads();
  float acc = 0.f;
  for (int k = 0; k < DIM; ++k) acc = fmaf(ol[k], Wo[(size_t)k * DIM + n], acc);
  outp[(size_t)b * DCAT + n] = acc + bo[n];     // caller pre-offsets outp by l*DIM
}

// ---------------- 10) MLP1: z = gelu(hcat @ W1 + b1), W1 read once ----------
__global__ void mlp1_kernel(const float* __restrict__ hcat, const float* __restrict__ W1,
                            const float* __restrict__ b1, float* __restrict__ z) {
  const int n = blockIdx.x * 256 + threadIdx.x;       // 0..IDIM-1
  __shared__ float hl[BATCH][256];
  float acc[BATCH];
#pragma unroll
  for (int b = 0; b < BATCH; ++b) acc[b] = 0.f;
  for (int k0 = 0; k0 < DCAT; k0 += 256) {
    __syncthreads();
    for (int i = threadIdx.x; i < BATCH * 256; i += 256)
      hl[i >> 8][i & 255] = hcat[(size_t)(i >> 8) * DCAT + k0 + (i & 255)];
    __syncthreads();
    for (int kk = 0; kk < 256; ++kk) {
      const float w = W1[(size_t)(k0 + kk) * IDIM + n];
#pragma unroll
      for (int b = 0; b < BATCH; ++b) acc[b] = fmaf(hl[b][kk], w, acc[b]);
    }
  }
#pragma unroll
  for (int b = 0; b < BATCH; ++b) {
    const float x = acc[b] + b1[n];
    z[(size_t)b * IDIM + n] = 0.5f * x * (1.f + erff(x * 0.70710678118654752440f));
  }
}

// ---------------- 11) MLP2 + residual: h2 = hcat + z @ W2 + b2 --------------
__global__ void mlp2_kernel(const float* __restrict__ z, const float* __restrict__ W2,
                            const float* __restrict__ b2, const float* __restrict__ hcat,
                            float* __restrict__ h2) {
  const int n = blockIdx.x * 256 + threadIdx.x;       // 0..DCAT-1
  __shared__ float zl[BATCH][256];
  float acc[BATCH];
#pragma unroll
  for (int b = 0; b < BATCH; ++b) acc[b] = 0.f;
  for (int k0 = 0; k0 < IDIM; k0 += 256) {
    __syncthreads();
    for (int i = threadIdx.x; i < BATCH * 256; i += 256)
      zl[i >> 8][i & 255] = z[(size_t)(i >> 8) * IDIM + k0 + (i & 255)];
    __syncthreads();
    for (int kk = 0; kk < 256; ++kk) {
      const float w = W2[(size_t)(k0 + kk) * DCAT + n];
#pragma unroll
      for (int b = 0; b < BATCH; ++b) acc[b] = fmaf(zl[b][kk], w, acc[b]);
    }
  }
#pragma unroll
  for (int b = 0; b < BATCH; ++b)
    h2[(size_t)b * DCAT + n] = hcat[(size_t)b * DCAT + n] + acc[b] + b2[n];
}

// ---------------- 12) logits: out[b,j] = h2[b,:] . Wl[:,j] + bl[j] ----------
__global__ void final_kernel(const float* __restrict__ h2, const float* __restrict__ Wl,
                             const float* __restrict__ bl, float* __restrict__ out) {
  const int b = blockIdx.x >> 1, j = blockIdx.x & 1;
  __shared__ float red[256];
  float acc = 0.f;
  for (int k = threadIdx.x; k < DCAT; k += 256)
    acc = fmaf(h2[(size_t)b * DCAT + k], Wl[(size_t)k * 2 + j], acc);
  red[threadIdx.x] = acc; __syncthreads();
  for (int st = 128; st > 0; st >>= 1) {
    if (threadIdx.x < st) red[threadIdx.x] += red[threadIdx.x + st];
    __syncthreads();
  }
  if (threadIdx.x == 0) out[b * 2 + j] = red[0] + bl[j];
}

// ---------------- launch ----------------
extern "C" void kernel_launch(void* const* d_in, const int* in_sizes, int n_in,
                              void* d_out, int out_size, void* d_ws, size_t ws_size,
                              hipStream_t stream) {
  (void)in_sizes; (void)n_in; (void)out_size; (void)ws_size;
  const float* hs[2] = { (const float*)d_in[0], (const float*)d_in[1] };
  const int*   ids  = (const int*)d_in[2];
  const float* ln_g = (const float*)d_in[3];
  const float* ln_b = (const float*)d_in[4];
  const float* Wq   = (const float*)d_in[5];
  const float* Wk   = (const float*)d_in[6];
  const float* Wv   = (const float*)d_in[7];
  const float* Wo   = (const float*)d_in[8];
  const float* bo   = (const float*)d_in[9];
  const float* W1   = (const float*)d_in[10];
  const float* b1   = (const float*)d_in[11];
  const float* W2   = (const float*)d_in[12];
  const float* b2   = (const float*)d_in[13];
  const float* Wl   = (const float*)d_in[14];
  const float* bl   = (const float*)d_in[15];
  float* out = (float*)d_out;

  // workspace carve-up (256B aligned slices)
  char* ws = (char*)d_ws;
  size_t off = 0;
  auto carve = [&](size_t bytes) -> void* {
    void* p = ws + off;
    off += (bytes + 255) & ~(size_t)255;
    return p;
  };
  __bf16* h_bf  = (__bf16*)carve((size_t)MTOK * DIM * 2);   //  64 MB (per-layer reuse)
  float*  Kf    = (float*) carve((size_t)MTOK * DIM * 4);   // 128 MB
  float*  Vf    = (float*) carve((size_t)MTOK * DIM * 4);   // 128 MB
  __bf16* WkP   = (__bf16*)carve((size_t)DIM * DIM * 2);    //   8 MB
  __bf16* WvP   = (__bf16*)carve((size_t)DIM * DIM * 2);    //   8 MB
  int*    idx   = (int*)   carve(BATCH * 4);
  float*  q     = (float*) carve((size_t)BATCH * DIM * 4);
  float*  sc    = (float*) carve((size_t)BATCH * NHEAD * SEQ * 4);
  float*  o     = (float*) carve((size_t)BATCH * DIM * 4);
  float*  hcat  = (float*) carve((size_t)BATCH * DCAT * 4);
  float*  z     = (float*) carve((size_t)BATCH * IDIM * 4);
  float*  h2    = (float*) carve((size_t)BATCH * DCAT * 4);

  idx_kernel<<<BATCH, 256, 0, stream>>>(ids, idx);

  for (int l = 0; l < LAYERS; ++l) {
    ln_kernel<<<MTOK, 256, 0, stream>>>(hs[l], ln_g + l * DIM, ln_b + l * DIM, h_bf);
    pack_w_kernel<<<(DIM * DIM) / 256, 256, 0, stream>>>(Wk + (size_t)l * DIM * DIM, WkP);
    pack_w_kernel<<<(DIM * DIM) / 256, 256, 0, stream>>>(Wv + (size_t)l * DIM * DIM, WvP);
    gemm_bf16_kernel<<<(MTOK / 16) * (DIM / 128) / 8, 256, 0, stream>>>(h_bf, WkP, Kf);
    gemm_bf16_kernel<<<(MTOK / 16) * (DIM / 128) / 8, 256, 0, stream>>>(h_bf, WvP, Vf);
    q_kernel<<<dim3(DIM / 256, BATCH), 256, 0, stream>>>(h_bf, idx, Wq + (size_t)l * DIM * DIM, q);
    scores_kernel<<<dim3(SEQ / 256, BATCH * NHEAD), 256, 0, stream>>>(q, Kf, sc);
    softmax_kernel<<<BATCH * NHEAD, 256, 0, stream>>>(sc);
    av_kernel<<<BATCH * NHEAD, HDIM, 0, stream>>>(sc, Vf, o);
    outproj_kernel<<<dim3(DIM / 256, BATCH), 256, 0, stream>>>(
        o, Wo + (size_t)l * DIM * DIM, bo + l * DIM, hcat + l * DIM);
  }

  mlp1_kernel<<<IDIM / 256, 256, 0, stream>>>(hcat, W1, b1, z);
  mlp2_kernel<<<DCAT / 256, 256, 0, stream>>>(z, W2, b2, hcat, h2);
  final_kernel<<<BATCH * 2, 256, 0, stream>>>(h2, Wl, bl, out);
}
